// Pixel_54571854463051
// MI455X (gfx1250) — matrix operand order repaired and checked by reading, not compiled, not run on hardware
//
#include <hip/hip_runtime.h>
#include <hip/hip_bf16.h>

// CDNA5 / gfx1250, wave32 only.
typedef __attribute__((ext_vector_type(16))) _Float16 v16h;
typedef __attribute__((ext_vector_type(8)))  _Float16 v8h;
typedef __attribute__((ext_vector_type(8)))  float    v8f;

#define NROW   4096          // b*M rows
#define NCOL   4096          // N (row length) = 64*64 image
#define SIDE   64
#define K_TOP  256
#define TPB    256           // 8 waves
#define EPT    (NCOL / TPB)  // 16 elements per thread

// ---- WMMA fragment helpers (ISA 7.12.2 layouts, wave32) -------------------
// All operand matrices are stored so every lane reads CONTIGUOUS f16 runs.

// A fragment: 16x32 f16 tile from a row-major matrix.
// lanes 0-15: M=lane, K={0..7,16..23}; lanes 16-31: M=lane-16, K={8..15,24..31}.
__device__ inline v16h load_frag_a(const _Float16* lds, int row0, int col0, int ld, int lane) {
    const int m     = lane & 15;
    const int khalf = (lane < 16) ? 0 : 8;
    const _Float16* p = lds + (row0 + m) * ld + col0 + khalf;
    v8h lo = *(const v8h*)(p);        // K = khalf .. khalf+7      (16B)
    v8h hi = *(const v8h*)(p + 16);   // K = 16+khalf .. 16+khalf+7 (16B)
    return __builtin_shufflevector(lo, hi, 0,1,2,3,4,5,6,7,8,9,10,11,12,13,14,15);
}

// B fragment: 32x16 f16 tile, matrix stored COLUMN-major (ldsT[n*ld + k]).
// lanes 0-15: N=lane, K={0..7,16..23}; lanes 16-31: N=lane-16, K={8..15,24..31}.
__device__ inline v16h load_frag_bT(const _Float16* ldsT, int k0, int col0, int ld, int lane) {
    const int n     = lane & 15;
    const int kbase = (lane < 16) ? 0 : 8;
    const _Float16* p = ldsT + (col0 + n) * ld + k0 + kbase;
    v8h lo = *(const v8h*)(p);
    v8h hi = *(const v8h*)(p + 16);
    return __builtin_shufflevector(lo, hi, 0,1,2,3,4,5,6,7,8,9,10,11,12,13,14,15);
}

// Store a C/D tile of Xᵀ into row-major X as f16: element (M,N) of Xᵀ goes to
// X[N][M] -> per lane (fixed N) the 8 M-values are contiguous -> one b128 store.
__device__ inline void store_cT_f16(_Float16* dstX, int row0, int col0, int ld,
                                    const v8f& c, int lane) {
    const int n  = lane & 15;
    const int mb = (lane < 16) ? 0 : 8;
    v8h h;
#pragma unroll
    for (int r = 0; r < 8; ++r) h[r] = (_Float16)c[r];
    *(v8h*)(dstX + (col0 + n) * ld + row0 + mb) = h;
}

// ---------------------------------------------------------------------------

__global__ void __launch_bounds__(TPB)
pixel_topk_dilate_kernel(const float* __restrict__ param,
                         const float* __restrict__ scalar,
                         const float* __restrict__ u,
                         float* __restrict__ out)
{
    __shared__ unsigned  keys[NCOL];          // 16 KB keys; reused as f32 T2
    __shared__ _Float16  imgT[SIDE * SIDE];   // 8 KB mask, TRANSPOSED
    __shared__ _Float16  band[SIDE * SIDE];   // 8 KB tridiagonal ones (symmetric)
    __shared__ _Float16  t1  [SIDE * SIDE];   // 8 KB T1 = band*img, row-major
    __shared__ unsigned  hist[256];
    __shared__ unsigned  sh_prefix, sh_k, sh_cnt_gt, sh_tie;
    __shared__ float     sh_sumsq;

    const int row  = blockIdx.x;       // 0..4095 (= b*M)
    const int m    = row & 127;        // param row (param is [1,128,4096])
    const int tid  = threadIdx.x;
    const int lane = tid & 31;
    const int wave = tid >> 5;

    if (tid == 0) { sh_prefix = 0u; sh_k = K_TOP; sh_cnt_gt = 0u; sh_sumsq = 0.0f; }

    // Band matrix: band[i][j] = 1 iff |i-j| <= 1 (separable 3x3 ones kernel).
    for (int idx = tid; idx < SIDE * SIDE; idx += TPB) {
        int d = (idx >> 6) - (idx & 63);
        band[idx] = (_Float16)((d >= -1 && d <= 1) ? 1.0f : 0.0f);
    }

    // Keys: param + gumbel/1000 (softmax is monotone -> same top-k order).
    const float* urow = u + (size_t)row * NCOL;
    const float* prow = param + (size_t)m * NCOL;
#pragma unroll
    for (int e = 0; e < EPT; ++e) {
        int n = tid + e * TPB;                       // coalesced
        float g   = -logf(-logf(urow[n]));
        float key = prow[n] + g * 1e-3f;
        unsigned bits = __float_as_uint(key);
        // order-preserving float -> uint (descending top-k = largest uints)
        keys[n] = bits ^ ((unsigned)((int)bits >> 31) | 0x80000000u);
    }
    __syncthreads();

    // ---- exact 256th-largest via 4-pass byte radix select ----
    for (int pass = 0; pass < 4; ++pass) {
        const int shift = 24 - pass * 8;
        const unsigned pmask = (pass == 0) ? 0u : (0xFFFFFFFFu << (shift + 8));
        hist[tid] = 0u;
        __syncthreads();
        const unsigned prefix = sh_prefix;
#pragma unroll
        for (int e = 0; e < EPT; ++e) {
            unsigned s = keys[tid + e * TPB];
            if ((s & pmask) == prefix)
                atomicAdd(&hist[(s >> shift) & 0xFFu], 1u);
        }
        __syncthreads();
        if (tid == 0) {
            unsigned k = sh_k, cum = 0u;
            int digit = 0;
            for (int b2 = 255; b2 >= 0; --b2) {
                unsigned c = hist[b2];
                if (cum + c >= k) { digit = b2; sh_k = k - cum; break; }
                cum += c;
            }
            sh_prefix = prefix | ((unsigned)digit << shift);
        }
        __syncthreads();
    }
    const unsigned thresh = sh_prefix;

    // count strictly-greater, then tie budget
    unsigned cnt = 0;
#pragma unroll
    for (int e = 0; e < EPT; ++e)
        cnt += (keys[tid + e * TPB] > thresh) ? 1u : 0u;
    atomicAdd(&sh_cnt_gt, cnt);
    __syncthreads();
    if (tid == 0) sh_tie = K_TOP - sh_cnt_gt;
    __syncthreads();

    // binary mask, written TRANSPOSED (exactly K_TOP ones): imgT[j][i]=mask[i][j]
#pragma unroll
    for (int e = 0; e < EPT; ++e) {
        int n = tid + e * TPB;
        unsigned s = keys[n];
        float v = 0.0f;
        if (s > thresh) v = 1.0f;
        else if (s == thresh) {
            unsigned old = atomicSub(&sh_tie, 1u);
            if ((int)old > 0) v = 1.0f;
        }
        imgT[(n & 63) * SIDE + (n >> 6)] = (_Float16)v;
    }
    __syncthreads();

    // ---- stage 1: T1ᵀ = imgᵀ·band  (== (band·img)ᵀ, band symmetric) ----
    // A = imgT (row-major), B = band (symmetric: row-major == column-major).
    // C tiles of T1ᵀ stored into row-major t1 via the contiguous-store trick.
#pragma unroll
    for (int tt = 0; tt < 2; ++tt) {
        const int t  = wave * 2 + tt;        // 16 tiles, 2 per wave
        const int ti = t >> 2, tj = t & 3;
        v8f acc = {};
#pragma unroll
        for (int kb = 0; kb < 2; ++kb) {
            v16h a = load_frag_a (imgT, ti * 16, kb * 32, SIDE, lane);
            v16h b = load_frag_bT(band, kb * 32, tj * 16, SIDE, lane);
            acc = __builtin_amdgcn_wmma_f32_16x16x32_f16(false, a, false, b,
                                                         (short)0, acc, false, false);
        }
        store_cT_f16(t1, ti * 16, tj * 16, SIDE, acc, lane);
    }
    __syncthreads();

    // ---- stage 2: T2 = T1·band; clamp to [0,1]; stage through LDS ----
    float* t2f = (float*)keys;   // keys are dead; reuse 16 KB as f32 T2
#pragma unroll
    for (int tt = 0; tt < 2; ++tt) {
        const int t  = wave * 2 + tt;
        const int ti = t >> 2, tj = t & 3;
        v8f acc = {};
#pragma unroll
        for (int kb = 0; kb < 2; ++kb) {
            v16h a = load_frag_a (t1,   ti * 16, kb * 32, SIDE, lane);
            v16h b = load_frag_bT(band, kb * 32, tj * 16, SIDE, lane);
            acc = __builtin_amdgcn_wmma_f32_16x16x32_f16(false, a, false, b,
                                                         (short)0, acc, false, false);
        }
        const int n_ = lane & 15;
        const int mb = (lane < 16) ? 0 : 8;
#pragma unroll
        for (int r = 0; r < 8; ++r) {
            float v = fminf(fmaxf(acc[r], 0.0f), 1.0f);   // clamp [0,1]
            t2f[(ti * 16 + mb + r) * SIDE + tj * 16 + n_] = v;
        }
    }
    __syncthreads();

    // ---- row L2 norm over all 4096 elements, then coalesced scaled writes ----
    float lsq = 0.0f;
#pragma unroll
    for (int e = 0; e < EPT; ++e) {
        float v = t2f[tid + e * TPB];
        lsq += v * v;
    }
    atomicAdd(&sh_sumsq, lsq);                 // ds_add_f32 reduction
    __syncthreads();

    const float sc = fmaxf(0.01f, scalar[0]) / (sqrtf(sh_sumsq) + 0.01f);

    float* orow = out + (size_t)row * NCOL;
#pragma unroll
    for (int e = 0; e < EPT; ++e) {
        int n = tid + e * TPB;                 // fully coalesced global stores
        orow[n] = t2f[n] * sc;
    }
}

extern "C" void kernel_launch(void* const* d_in, const int* in_sizes, int n_in,
                              void* d_out, int out_size, void* d_ws, size_t ws_size,
                              hipStream_t stream) {
    (void)in_sizes; (void)n_in; (void)d_ws; (void)ws_size; (void)out_size;
    const float* param  = (const float*)d_in[0];   // [1,128,4096] f32
    const float* scalar = (const float*)d_in[1];   // [1] f32
    const float* u      = (const float*)d_in[2];   // [32,128,4096] f32
    float*       out    = (float*)d_out;           // [32,128,4096] f32

    pixel_topk_dilate_kernel<<<dim3(NROW), dim3(TPB), 0, stream>>>(param, scalar, u, out);
}
